// GroupCRouter_78288663872328
// MI455X (gfx1250) — matrix operand order repaired and checked by reading, not compiled, not run on hardware
//
#include <hip/hip_runtime.h>
#include <hip/hip_bf16.h>
#include <math.h>

typedef __bf16 bf16_t;
typedef __attribute__((ext_vector_type(16))) __bf16 v16bf;
typedef __attribute__((ext_vector_type(8)))  float  v8f;
typedef __attribute__((ext_vector_type(8)))  unsigned int v8u;

#define D_DIM 1024
#define H_DIM 256
#define E_EXP 6
#define N_SEQ 4096
#define TOK_PER_BLK 64
#define KCHUNK 64
#define HS_STRIDE (H_DIM + 8)

// ---------------------------------------------------------------------------
// Pre-pack W1 (1024x256 f32, row-major [k][col]) into bf16 WMMA B-fragments.
// Fragment index = ((ct*32 + kc)*32 + lane), 16 bf16 each.
// ISA 16-bit B (32x16) layout: lane -> col = lane&15, K offset +8 for lanes
// 16..31, VGPR j holds K pair {2j, 2j+1} (j<4) or {8+2j, 9+2j} (j>=4).
// ---------------------------------------------------------------------------
__global__ void prepack_w1(const float* __restrict__ W1, bf16_t* __restrict__ Bp) {
  int tid = blockIdx.x * blockDim.x + threadIdx.x;
  if (tid >= 16 * 32 * 32) return;          // ct(16) * kc(32) * lane(32)
  int lane = tid & 31;
  int kc   = (tid >> 5) & 31;
  int ct   = tid >> 10;
  int col  = ct * 16 + (lane & 15);
  int koff = (lane & 16) ? 8 : 0;
  v8u frag;
#pragma unroll
  for (int j = 0; j < 8; ++j) {
    int k = kc * 32 + koff + (j < 4 ? 2 * j : 8 + 2 * j);
    bf16_t lo = (bf16_t)W1[(size_t)k * H_DIM + col];
    bf16_t hi = (bf16_t)W1[(size_t)(k + 1) * H_DIM + col];
    unsigned short us_lo, us_hi;
    __builtin_memcpy(&us_lo, &lo, 2);
    __builtin_memcpy(&us_hi, &hi, 2);
    frag[j] = (unsigned int)us_lo | ((unsigned int)us_hi << 16);
  }
  *(v8u*)(Bp + (size_t)tid * 16) = frag;
}

// ---------------------------------------------------------------------------
// Main router kernel: 256 threads (8 waves = 2 Mwaves x 4 Nwaves),
// 64 tokens per block. Each wave: 2 M-tiles x 4 N-tiles -> 8 accumulators.
// ---------------------------------------------------------------------------
__global__ __launch_bounds__(256) void router_kernel(
    const float* __restrict__ tokens, const int* __restrict__ token_types,
    const int* __restrict__ t_arr, const bf16_t* __restrict__ W1p,
    const float* __restrict__ b1, const float* __restrict__ W2,
    const float* __restrict__ b2, const float* __restrict__ base_assign,
    float* __restrict__ out_dispatch, float* __restrict__ out_combine) {
  __shared__ bf16_t As[TOK_PER_BLK][KCHUNK];       // 8 KB
  __shared__ bf16_t hs[TOK_PER_BLK][HS_STRIDE];    // 33 KB (gelu(h) in bf16)
  __shared__ float  w2s[H_DIM * E_EXP];            // 6 KB
  __shared__ float  b2s[E_EXP];
  __shared__ float  lg[TOK_PER_BLK][E_EXP];        // 1.5 KB

  const int tid  = threadIdx.x;
  const int lane = tid & 31;
  const int wave = tid >> 5;
  const int mw   = wave >> 2;       // 0..1 : token rows [mw*32, mw*32+32)
  const int nw   = wave & 3;        // 0..3 : h columns  [nw*64, nw*64+64)
  const int rowbase = blockIdx.x * TOK_PER_BLK;

  // stage W2/b2 into LDS (visible after the first in-loop barrier)
  for (int i = tid; i < H_DIM * E_EXP; i += 256) w2s[i] = W2[i];
  if (tid < E_EXP) b2s[tid] = b2[tid];

  v8f acc[2][4];
#pragma unroll
  for (int mi = 0; mi < 2; ++mi)
#pragma unroll
    for (int ni = 0; ni < 4; ++ni) acc[mi][ni] = {};

  const int arow = tid >> 2;           // 0..63 (token row within tile)
  const int agrp = tid & 3;            // 0..3  (16 floats each)
  const float* tokrow = tokens + (size_t)(rowbase + arow) * D_DIM + agrp * 16;

  const int frow  = lane & 15;
  const int fkoff = (lane & 16) ? 8 : 0;  // A-layout lane-group K offset
  const int ctbase = nw * 4;

  for (int k0 = 0; k0 < D_DIM; k0 += KCHUNK) {
    float4 f0 = *(const float4*)(tokrow + k0);
    float4 f1 = *(const float4*)(tokrow + k0 + 4);
    float4 f2 = *(const float4*)(tokrow + k0 + 8);
    float4 f3 = *(const float4*)(tokrow + k0 + 12);
    __syncthreads();  // previous chunk fully consumed
    bf16_t* dst = &As[arow][agrp * 16];
    dst[0]  = (bf16_t)f0.x; dst[1]  = (bf16_t)f0.y;
    dst[2]  = (bf16_t)f0.z; dst[3]  = (bf16_t)f0.w;
    dst[4]  = (bf16_t)f1.x; dst[5]  = (bf16_t)f1.y;
    dst[6]  = (bf16_t)f1.z; dst[7]  = (bf16_t)f1.w;
    dst[8]  = (bf16_t)f2.x; dst[9]  = (bf16_t)f2.y;
    dst[10] = (bf16_t)f2.z; dst[11] = (bf16_t)f2.w;
    dst[12] = (bf16_t)f3.x; dst[13] = (bf16_t)f3.y;
    dst[14] = (bf16_t)f3.z; dst[15] = (bf16_t)f3.w;
    __syncthreads();  // chunk staged

#pragma unroll
    for (int ks = 0; ks < 2; ++ks) {
      // A fragments (16x32 bf16) per ISA layout:
      // lanes 0-15: V0..3 -> K 0..7, V4..7 -> K 16..23; lanes 16-31: +8
      v16bf a[2];
#pragma unroll
      for (int mi = 0; mi < 2; ++mi) {
        int row = mw * 32 + mi * 16 + frow;
        v8u au;
#pragma unroll
        for (int j = 0; j < 8; ++j) {
          int kb = (j < 4 ? 2 * j : 8 + 2 * j) + fkoff + ks * 32;
          au[j] = *(const unsigned int*)&As[row][kb];
        }
        a[mi] = __builtin_bit_cast(v16bf, au);
      }

      int kc = (k0 >> 5) + ks;  // global K-chunk-of-32 index, 0..31
      const v16bf* Bp = (const v16bf*)W1p;
      v16bf bv[4];
#pragma unroll
      for (int ni = 0; ni < 4; ++ni)
        bv[ni] = Bp[(size_t)(((ctbase + ni) * 32 + kc) * 32) + lane];

#pragma unroll
      for (int mi = 0; mi < 2; ++mi) {
#pragma unroll
        for (int ni = 0; ni < 4; ++ni) {
          acc[mi][ni] = __builtin_amdgcn_wmma_f32_16x16x32_bf16(
              false, a[mi], false, bv[ni], (short)0, acc[mi][ni], false, false);
        }
      }
    }
  }

  // Epilogue: h = gelu(acc + b1) -> LDS (bf16).
  // C/D f32 layout: VGPR r, lanes 0-15 -> M=r, lanes 16-31 -> M=r+8; N=lane&15.
  {
    const int coln   = lane & 15;
    const int rowoff = ((lane >> 4) & 1) * 8;
#pragma unroll
    for (int mi = 0; mi < 2; ++mi) {
#pragma unroll
      for (int ni = 0; ni < 4; ++ni) {
        int col = (ctbase + ni) * 16 + coln;
        float bb = b1[col];
#pragma unroll
        for (int r = 0; r < 8; ++r) {
          int mrow = mw * 32 + mi * 16 + r + rowoff;
          float x = acc[mi][ni][r] + bb;
          float g = 0.5f * x * (1.0f + erff(x * 0.70710678118654752f));
          hs[mrow][col] = (bf16_t)g;
        }
      }
    }
  }
  __syncthreads();

  // Tiny second GEMM: logits = (h @ W2 + b2) / TEMP, 64 tokens x 6 experts.
  for (int i = tid; i < TOK_PER_BLK * E_EXP; i += 256) {
    int tok = i / E_EXP, e = i % E_EXP;
    float s = b2s[e];
#pragma unroll 8
    for (int k = 0; k < H_DIM; ++k) s += (float)hs[tok][k] * w2s[k * E_EXP + e];
    lg[tok][e] = s * 10.0f;  // 1/TEMP
  }
  __syncthreads();

  // Per-token tail: blend, floor, top-1, hard cap, normalize.
  if (tid < TOK_PER_BLK) {
    int g = rowbase + tid;
    int b = g >> 12;  // N_SEQ = 4096
    int ttype = token_types[g];
    float tn  = (float)t_arr[b] * 1e-3f;
    float cap = 0.5f + 1.1f * tn;  // CAP_LOW + (CAP_HIGH+CAP_LOW)*t_norm (faithful)
    bool unk = (ttype == 5);

    float p[E_EXP];
#pragma unroll
    for (int e = 0; e < E_EXP; ++e) {
      float base = base_assign[ttype * E_EXP + e];
      float v = unk ? lg[tid][e] : base;
      p[e] = 0.775f * v + 0.0375f;  // (1-alpha)*p + alpha/E, alpha = 0.225
    }
    // top-1 (first index wins ties, matching lax.top_k)
    int idx = 0; float best = p[0];
#pragma unroll
    for (int e = 1; e < E_EXP; ++e)
      if (p[e] > best) { best = p[e]; idx = e; }

    float d[E_EXP];
#pragma unroll
    for (int e = 0; e < E_EXP; ++e) d[e] = (e == idx) ? best : 0.0f;

    // hard cap with headroom redistribution
    float ex_sum = 0.0f, capped[E_EXP], hr[E_EXP], hr_sum = 0.0f;
#pragma unroll
    for (int e = 0; e < E_EXP; ++e) {
      float ex = fmaxf(d[e] - cap, 0.0f);
      ex_sum += ex;
      capped[e] = d[e] - ex;
    }
#pragma unroll
    for (int e = 0; e < E_EXP; ++e) {
      hr[e] = fmaxf(cap - capped[e], 0.0f);
      hr_sum += hr[e];
    }
    hr_sum = fmaxf(hr_sum, 1e-8f);
    float disp[E_EXP], dsum = 0.0f;
#pragma unroll
    for (int e = 0; e < E_EXP; ++e) {
      disp[e] = capped[e] + ex_sum * (hr[e] / hr_sum);
      dsum += disp[e];
    }
    float inv = 1.0f / (dsum + 1e-8f);
    size_t off = (size_t)g * E_EXP;
#pragma unroll
    for (int e = 0; e < E_EXP; ++e) {
      out_dispatch[off + e] = disp[e];
      out_combine[off + e]  = disp[e] * inv;
    }
  }
}

extern "C" void kernel_launch(void* const* d_in, const int* in_sizes, int n_in,
                              void* d_out, int out_size, void* d_ws, size_t ws_size,
                              hipStream_t stream) {
  (void)in_sizes; (void)n_in; (void)out_size; (void)ws_size;
  const float* tokens      = (const float*)d_in[0];
  const int*   token_types = (const int*)d_in[1];
  const int*   t_arr       = (const int*)d_in[2];
  const float* W1          = (const float*)d_in[3];
  const float* b1          = (const float*)d_in[4];
  const float* W2          = (const float*)d_in[5];
  const float* b2          = (const float*)d_in[6];
  const float* base        = (const float*)d_in[7];

  bf16_t* W1p = (bf16_t*)d_ws;  // 16*32*32 fragments * 16 bf16 = 512 KB

  float* out_dispatch = (float*)d_out;
  float* out_combine  = out_dispatch + (size_t)8 * N_SEQ * E_EXP;

  prepack_w1<<<64, 256, 0, stream>>>(W1, W1p);

  int total_tokens = 8 * N_SEQ;
  int blocks = total_tokens / TOK_PER_BLK;  // 512
  router_kernel<<<blocks, 256, 0, stream>>>(tokens, token_types, t_arr, W1p,
                                            b1, W2, b2, base,
                                            out_dispatch, out_combine);
}
